// MambaG2G2_31318901522696
// MI455X (gfx1250) — matrix operand order repaired and checked by reading, not compiled, not run on hardware
//
#include <hip/hip_runtime.h>
#include <hip/hip_bf16.h>
#include <math.h>

// ---------------- problem constants ----------------
#define NN       96      // N (nodes / batch)
#define TT       160     // T (sequence length)
#define DD       96      // D (feature dim)
#define D_STATE  64
#define D_CONV   4
#define HEADDIM  32
#define NGROUPS  6
#define D_INNER  192     // 2*D
#define NHEADS   6       // D_INNER / HEADDIM
#define EMBD     24
#define DOUT     64
#define D_IN_PROJ 1158   // 2*D_INNER + 2*NGROUPS*D_STATE + NHEADS
#define CONV_DIM  960    // D_INNER + 2*NGROUPS*D_STATE
#define LDZ       1216   // padded in_proj output stride (19*64)
#define ROWS      (TT*NN)   // 15360 rows, row r = t*96 + n

typedef __attribute__((ext_vector_type(16))) _Float16 v16h;
typedef __attribute__((ext_vector_type(8)))  _Float16 v8h;
typedef __attribute__((ext_vector_type(8)))  float    v8f;

__device__ __forceinline__ float siluf(float v) { return v / (1.0f + __expf(-v)); }

// ====================================================================
// Weight pack: f32 B[K x Nb] -> f16 BP[col][K] (transposed), cols padded
// with zeros up to gridDim.x.
// ====================================================================
__global__ void pack_w_kernel(const float* __restrict__ B,
                              _Float16* __restrict__ BP,
                              int K, int Nb)
{
    const int col = blockIdx.x;
    for (int k = threadIdx.x; k < K; k += blockDim.x) {
        float v = (col < Nb) ? B[(size_t)k * Nb + col] : 0.0f;
        BP[(size_t)col * K + k] = (_Float16)v;
    }
}

// ====================================================================
// Kernel 1: aggregation.  One block per t.  Emits gin in f16 (GEMM A
// operand) and xs in f32 (residual for GEMM2).
// ====================================================================
__global__ void agg_kernel(const float* __restrict__ input,
                           const int*   __restrict__ hem,
                           const int*   __restrict__ subnet,
                           const float* __restrict__ hem_emb,
                           const float* __restrict__ subnet_emb,
                           float* __restrict__ xs_rows,
                           _Float16* __restrict__ gin16)
{
    const int t = blockIdx.x;
    __shared__ float sx[NN * 48];   // xs[t, i, 48+c]
    __shared__ float sa[NN * 48];   // a[i, c]
    __shared__ float s1[48];

    for (int i = threadIdx.x; i < NN * 48; i += blockDim.x) {
        int n = i / 48, c = i % 48;
        sx[i] = input[((size_t)n * TT + t) * DD + 48 + c];
        sa[i] = (c < EMBD) ? hem_emb[hem[n] * EMBD + c]
                           : subnet_emb[subnet[n] * EMBD + (c - EMBD)];
    }
    __syncthreads();

    if (threadIdx.x < 48) {
        int c = threadIdx.x;
        float s = 0.0f;
        for (int i = 0; i < NN; ++i)
            s += fmaxf(input[((size_t)i * TT + t) * DD + c] + sa[i * 48 + c], 0.0f);
        s1[c] = s;
    }

    for (int o = threadIdx.x; o < NN * 48; o += blockDim.x) {
        int j = o / 48, c = o % 48;
        float aj = sa[j * 48 + c];
        float sum = 0.0f;
        for (int i = 0; i < NN; ++i)
            sum += fmaxf(sx[i * 48 + c] + aj, 0.0f);
        float xv = sx[j * 48 + c];
        size_t r = (size_t)t * NN + j;
        xs_rows[r * DD + 48 + c] = xv;
        gin16[r * DD + 48 + c] = (_Float16)(xv + sum);
    }
    __syncthreads();

    for (int o = threadIdx.x; o < NN * 48; o += blockDim.x) {
        int j = o / 48, c = o % 48;
        float xv = input[((size_t)j * TT + t) * DD + c];
        size_t r = (size_t)t * NN + j;
        xs_rows[r * DD + c] = xv;
        gin16[r * DD + c] = (_Float16)(xv + s1[c]);
    }
}

// ====================================================================
// WMMA GEMM (register-direct, fully compile-time specialized):
//   C[M x LDC] = act(A16[M x K] @ BP + bias + resid)
//   A16: f16 row-major, row stride = K (16B-aligned fragments)
//   BP : f16 packed [col][K] (transposed, padded cols) -> each lane's
//        B fragment is two 16B runs.
//   block = 128 threads (4 wave32); block tile 16 rows x 64 cols.
//   No LDS, no barriers; K loop fully unrolled; branch-free epilogue.
// ====================================================================
template <int ACT, int K, bool OUT16, bool HAS_BIAS, bool HAS_RESID,
          int LDC, int NBSTORE>
__global__ void gemm_wmma(const _Float16* __restrict__ A16,
                          const _Float16* __restrict__ BP,
                          const float* __restrict__ bias,
                          const float* __restrict__ resid,
                          float* __restrict__ Cf,
                          _Float16* __restrict__ Ch)
{
    const int tid  = threadIdx.x;
    const int wave = tid >> 5;
    const int lane = tid & 31;
    const int hi   = lane >> 4;
    const int lrow = lane & 15;
    const int rowBase = blockIdx.x * 16;
    const int col     = blockIdx.y * 64 + wave * 16 + lrow;

    const _Float16* arow = A16 + (size_t)(rowBase + lrow) * K;
    const _Float16* bcol = BP  + (size_t)col * K;

    v8f acc = {};
    #pragma unroll
    for (int kk = 0; kk < K; kk += 32) {
        // A fragment: K idx = {hi*8 + j | j<8} U {16 + hi*8 + j-8}
        v8h a0 = *(const v8h*)(arow + kk + hi * 8);
        v8h a1 = *(const v8h*)(arow + kk + 16 + hi * 8);
        v16h af = __builtin_shufflevector(a0, a1,
                    0,1,2,3,4,5,6,7,8,9,10,11,12,13,14,15);
        // B fragment: K idx = {hi*8 + j | j<8} U {16 + hi*8 + j-8}
        v8h b0 = *(const v8h*)(bcol + kk + hi * 8);
        v8h b1 = *(const v8h*)(bcol + kk + 16 + hi * 8);
        v16h bf = __builtin_shufflevector(b0, b1,
                    0,1,2,3,4,5,6,7,8,9,10,11,12,13,14,15);
        acc = __builtin_amdgcn_wmma_f32_16x16x32_f16(
                  false, af, false, bf, (short)0, acc, false, false);
    }

    // C/D layout: VGPR v holds M = v + hi*8, N = lane&15
    if (NBSTORE >= LDC || col < NBSTORE) {
        float bv = HAS_BIAS ? bias[col] : 0.0f;
        #pragma unroll
        for (int v = 0; v < 8; ++v) {
            int row = rowBase + hi * 8 + v;
            float val = acc[v] + bv;
            if (HAS_RESID) val += resid[(size_t)row * LDC + col];
            if (ACT == 1)  val = fmaxf(val, 0.0f);
            if (OUT16) Ch[(size_t)row * LDC + col] = (_Float16)val;
            else       Cf[(size_t)row * LDC + col] = val;
        }
    }
}

// ====================================================================
// Kernel: per-t BatchNorm over the node axis; emits f16 for GEMM3.
// ====================================================================
__global__ void bn_kernel(const float* __restrict__ h,
                          const float* __restrict__ bn_w,
                          const float* __restrict__ bn_b,
                          _Float16* __restrict__ h_bn16)
{
    const int t = blockIdx.x;
    const int d = threadIdx.x;
    if (d >= DD) return;
    float mu = 0.0f;
    for (int n = 0; n < NN; ++n) mu += h[((size_t)t * NN + n) * DD + d];
    mu *= (1.0f / NN);
    float var = 0.0f;
    for (int n = 0; n < NN; ++n) {
        float v = h[((size_t)t * NN + n) * DD + d] - mu;
        var += v * v;
    }
    var *= (1.0f / NN);
    float inv = rsqrtf(var + 1e-5f) * bn_w[d];
    float bb  = bn_b[d];
    for (int n = 0; n < NN; ++n) {
        size_t idx = ((size_t)t * NN + n) * DD + d;
        h_bn16[idx] = (_Float16)((h[idx] - mu) * inv + bb);
    }
}

// ====================================================================
// Kernel: causal depthwise conv (D_CONV=4) + SiLU + split + softplus(dt).
// ====================================================================
__global__ void conv_kernel(const float* __restrict__ zx,
                            const float* __restrict__ conv_w,
                            const float* __restrict__ conv_b,
                            const float* __restrict__ dt_bias,
                            float* __restrict__ x_ws,   // (n,t, 192)
                            float* __restrict__ B_ws,   // (n,t, 384)
                            float* __restrict__ C_ws,   // (n,t, 384)
                            float* __restrict__ dt_ws)  // (n,t, 6)
{
    const int n = blockIdx.x / TT;
    const int t = blockIdx.x % TT;
    const size_t ob = (size_t)n * TT + t;

    for (int c = threadIdx.x; c < CONV_DIM; c += blockDim.x) {
        float acc = conv_b[c];
        #pragma unroll
        for (int k = 0; k < D_CONV; ++k) {
            int tt = t - (D_CONV - 1) + k;
            if (tt >= 0)
                acc += conv_w[c * D_CONV + k] *
                       zx[((size_t)tt * NN + n) * LDZ + D_INNER + c];
        }
        float sc = siluf(acc);
        if (c < D_INNER)                     x_ws[ob * D_INNER + c] = sc;
        else if (c < D_INNER + NGROUPS * D_STATE)
                                             B_ws[ob * (NGROUPS * D_STATE) + (c - D_INNER)] = sc;
        else                                 C_ws[ob * (NGROUPS * D_STATE) + (c - D_INNER - NGROUPS * D_STATE)] = sc;
    }
    if (threadIdx.x < NHEADS) {
        int hh = threadIdx.x;
        float v = zx[((size_t)t * NN + n) * LDZ + (D_IN_PROJ - NHEADS) + hh] + dt_bias[hh];
        dt_ws[ob * NHEADS + hh] = (v > 20.0f) ? v : log1pf(__expf(v));
    }
}

// ====================================================================
// Kernel: sequential SSM scan.  One block per (batch n, head h);
// 256 threads hold the 32x64 state (8 f32 each).
// ====================================================================
__global__ void scan_kernel(const float* __restrict__ x_ws,
                            const float* __restrict__ B_ws,
                            const float* __restrict__ C_ws,
                            const float* __restrict__ dt_ws,
                            const float* __restrict__ A_log,
                            const float* __restrict__ Dparam,
                            float* __restrict__ y_ws)   // rows (t,n), 192 wide
{
    const int n = blockIdx.x / NHEADS;
    const int h = blockIdx.x % NHEADS;   // group == head (NGROUPS == NHEADS)
    const int tid = threadIdx.x;
    const int p  = tid >> 3;             // head-dim index 0..31
    const int so = (tid & 7) * 8;        // state-dim slice start

    __shared__ float sx[HEADDIM];
    __shared__ float sB[D_STATE];
    __shared__ float sC[D_STATE];
    __shared__ float sdt;

    const float Ah = -__expf(A_log[h]);
    const float Dp = Dparam[h];

    float st[8];
    #pragma unroll
    for (int i = 0; i < 8; ++i) st[i] = 0.0f;

    for (int t = 0; t < TT; ++t) {
        const size_t ob = (size_t)n * TT + t;
        if (tid < 32)                    sx[tid]       = x_ws[ob * D_INNER + h * HEADDIM + tid];
        else if (tid < 96)               sB[tid - 32]  = B_ws[ob * (NGROUPS * D_STATE) + h * D_STATE + (tid - 32)];
        else if (tid < 160)              sC[tid - 96]  = C_ws[ob * (NGROUPS * D_STATE) + h * D_STATE + (tid - 96)];
        else if (tid == 160)             sdt           = dt_ws[ob * NHEADS + h];
        __syncthreads();

        const float dtt  = sdt;
        const float dA   = __expf(dtt * Ah);
        const float xv   = sx[p];
        const float coef = dtt * xv;
        float acc = 0.0f;
        #pragma unroll
        for (int i = 0; i < 8; ++i) {
            float s = st[i] * dA + coef * sB[so + i];
            st[i] = s;
            acc += s * sC[so + i];
        }
        acc += __shfl_xor(acc, 1);
        acc += __shfl_xor(acc, 2);
        acc += __shfl_xor(acc, 4);
        if ((tid & 7) == 0)
            y_ws[((size_t)t * NN + n) * D_INNER + h * HEADDIM + p] = acc + Dp * xv;
        __syncthreads();
    }
}

// ====================================================================
// Kernel: gating + RMSNorm; emits f16 for GEMM4.
// ====================================================================
__global__ void gate_kernel(const float* __restrict__ y_ws,
                            const float* __restrict__ zx,   // zg = cols 0:192
                            const float* __restrict__ norm_w,
                            _Float16* __restrict__ ynorm16)
{
    const size_t r = blockIdx.x;
    const int d = threadIdx.x;
    __shared__ float red[8];
    __shared__ float stot;

    float val = 0.0f;
    if (d < D_INNER) {
        float zv = zx[r * LDZ + d];
        val = y_ws[r * D_INNER + d] * siluf(zv);
    }
    float sq = val * val;
    #pragma unroll
    for (int o = 16; o >= 1; o >>= 1) sq += __shfl_xor(sq, o);
    if ((threadIdx.x & 31) == 0) red[threadIdx.x >> 5] = sq;
    __syncthreads();
    if (threadIdx.x == 0) {
        float s = 0.0f;
        #pragma unroll
        for (int i = 0; i < 8; ++i) s += red[i];
        stot = s;
    }
    __syncthreads();
    float inv = rsqrtf(stot * (1.0f / D_INNER) + 1e-5f);
    if (d < D_INNER)
        ynorm16[r * D_INNER + d] = (_Float16)(val * inv * norm_w[d]);
}

// ====================================================================
// Kernel: mean over L, then out_fc (tanh+elu) and mu/sigma heads.
// ====================================================================
__global__ void head_kernel(const float* __restrict__ e_ws,     // rows (t,n), 96 wide
                            const float* __restrict__ out_fc_w,
                            const float* __restrict__ out_fc_b,
                            const float* __restrict__ mu_w,
                            const float* __restrict__ mu_b,
                            const float* __restrict__ sigma_w,
                            const float* __restrict__ sigma_b,
                            float* __restrict__ out)
{
    const int n = blockIdx.x;
    const int tid = threadIdx.x;
    __shared__ float em[DD];
    __shared__ float sx1[DD];

    if (tid < DD) {
        float s = 0.0f;
        for (int t = 0; t < TT; ++t)
            s += e_ws[((size_t)t * NN + n) * DD + tid];
        em[tid] = s * (1.0f / TT);
    }
    __syncthreads();
    if (tid < DD) {
        float s = out_fc_b[tid];
        for (int d = 0; d < DD; ++d) s += em[d] * out_fc_w[d * DD + tid];
        float th = tanhf(s);
        float el = (th > 0.0f) ? th : (__expf(th) - 1.0f);
        sx1[tid] = el;
        out[(size_t)n * DD + tid] = el;                     // x1
    }
    __syncthreads();
    if (tid < DOUT) {
        float m  = mu_b[tid];
        float sg = sigma_b[tid];
        for (int d = 0; d < DD; ++d) {
            m  += sx1[d] * mu_w[d * DOUT + tid];
            sg += sx1[d] * sigma_w[d * DOUT + tid];
        }
        out[NN * DD + (size_t)n * DOUT + tid] = m;          // mu
        float el = (sg > 0.0f) ? sg : (__expf(sg) - 1.0f);
        out[NN * DD + NN * DOUT + (size_t)n * DOUT + tid] = el + 1.0f + 1e-14f; // sigma
    }
}

// ====================================================================
extern "C" void kernel_launch(void* const* d_in, const int* in_sizes, int n_in,
                              void* d_out, int out_size, void* d_ws, size_t ws_size,
                              hipStream_t stream)
{
    (void)in_sizes; (void)n_in; (void)out_size; (void)ws_size;

    const float* input      = (const float*)d_in[0];
    const int*   hem        = (const int*)  d_in[1];
    const int*   subnet     = (const int*)  d_in[2];
    const float* gine_w1    = (const float*)d_in[3];
    const float* gine_b1    = (const float*)d_in[4];
    const float* gine_w2    = (const float*)d_in[5];
    const float* gine_b2    = (const float*)d_in[6];
    const float* bn_w       = (const float*)d_in[7];
    const float* bn_b       = (const float*)d_in[8];
    const float* hem_emb    = (const float*)d_in[9];
    const float* subnet_emb = (const float*)d_in[10];
    const float* in_proj    = (const float*)d_in[11];
    const float* conv_w     = (const float*)d_in[12];
    const float* conv_b     = (const float*)d_in[13];
    const float* dt_bias    = (const float*)d_in[14];
    const float* A_log      = (const float*)d_in[15];
    const float* Dparam     = (const float*)d_in[16];
    const float* norm_w     = (const float*)d_in[17];
    const float* out_proj   = (const float*)d_in[18];
    const float* out_fc_w   = (const float*)d_in[19];
    const float* out_fc_b   = (const float*)d_in[20];
    const float* mu_w       = (const float*)d_in[21];
    const float* mu_b       = (const float*)d_in[22];
    const float* sigma_w    = (const float*)d_in[23];
    const float* sigma_b    = (const float*)d_in[24];
    float* out = (float*)d_out;

    // ---- workspace layout ----
    float* ws = (float*)d_ws;
    const size_t R = (size_t)ROWS;
    float* xs_rows = ws;                      // R*96   f32
    float* h       = xs_rows + R * DD;        // R*96   f32
    float* zx      = h       + R * DD;        // R*1216 f32
    float* x_ws    = zx      + R * LDZ;       // R*192  f32
    float* B_ws    = x_ws    + R * D_INNER;   // R*384  f32
    float* C_ws    = B_ws    + R * 384;       // R*384  f32
    float* dt_ws   = C_ws    + R * 384;       // R*6    f32
    float* y_ws    = dt_ws   + R * NHEADS;    // R*192  f32
    float* e_ws    = y_ws    + R * D_INNER;   // R*96   f32
    float* f16base = e_ws    + R * DD;
    _Float16* gin16    = (_Float16*)f16base;            // R*96
    _Float16* mm1_16   = gin16    + R * DD;             // R*96
    _Float16* h_bn16   = mm1_16   + R * DD;             // R*96
    _Float16* ynorm16  = h_bn16   + R * DD;             // R*192
    _Float16* w1p      = ynorm16  + R * D_INNER;        // 128*96
    _Float16* w2p      = w1p      + 128 * DD;           // 128*96
    _Float16* inpp     = w2p      + 128 * DD;           // 1216*96
    _Float16* outpp    = inpp     + (size_t)LDZ * DD;   // 128*192

    // 0) pack weights: f32 [K][Nb] -> f16 [colPad][K]
    pack_w_kernel<<<128, 64, 0, stream>>>(gine_w1,  w1p,   DD,      DD);
    pack_w_kernel<<<128, 64, 0, stream>>>(gine_w2,  w2p,   DD,      DD);
    pack_w_kernel<<<LDZ, 64, 0, stream>>>(in_proj,  inpp,  DD,      D_IN_PROJ);
    pack_w_kernel<<<128, 64, 0, stream>>>(out_proj, outpp, D_INNER, DD);

    // 1) aggregation -> gin16, xs_rows
    agg_kernel<<<TT, 256, 0, stream>>>(input, hem, subnet, hem_emb, subnet_emb,
                                       xs_rows, gin16);

    // 2) mm1 = relu(gin @ gine_w1 + b1)   (f16 out)
    gemm_wmma<1, DD, true, true, false, DD, DD>
        <<<dim3(ROWS / 16, 2), 128, 0, stream>>>(
        gin16, w1p, gine_b1, nullptr, nullptr, mm1_16);

    // 3) h = mm1 @ gine_w2 + b2 + xs     (f32 out)
    gemm_wmma<0, DD, false, true, true, DD, DD>
        <<<dim3(ROWS / 16, 2), 128, 0, stream>>>(
        mm1_16, w2p, gine_b2, xs_rows, h, nullptr);

    // 4) per-t BatchNorm -> h_bn16
    bn_kernel<<<TT, 128, 0, stream>>>(h, bn_w, bn_b, h_bn16);

    // 5) zxbcdt = h_bn @ in_proj  (store all padded 1216 cols)
    gemm_wmma<0, DD, false, false, false, LDZ, LDZ>
        <<<dim3(ROWS / 16, 19), 128, 0, stream>>>(
        h_bn16, inpp, nullptr, nullptr, zx, nullptr);

    // 6) conv + silu + split + softplus(dt)
    conv_kernel<<<NN * TT, 256, 0, stream>>>(zx, conv_w, conv_b, dt_bias,
                                             x_ws, B_ws, C_ws, dt_ws);

    // 7) sequential scan over L
    scan_kernel<<<NN * NHEADS, 256, 0, stream>>>(x_ws, B_ws, C_ws, dt_ws,
                                                 A_log, Dparam, y_ws);

    // 8) gating + RMSNorm -> ynorm16
    gate_kernel<<<ROWS, 256, 0, stream>>>(y_ws, zx, norm_w, ynorm16);

    // 9) e = ynorm @ out_proj
    gemm_wmma<0, D_INNER, false, false, false, DD, DD>
        <<<dim3(ROWS / 16, 2), 128, 0, stream>>>(
        ynorm16, outpp, nullptr, nullptr, e_ws, nullptr);

    // 10) mean over L + heads -> x1, mu, sigma
    head_kernel<<<NN, 128, 0, stream>>>(e_ws, out_fc_w, out_fc_b,
                                        mu_w, mu_b, sigma_w, sigma_b, out);
}
